// Text2ImgCA_39049842655849
// MI455X (gfx1250) — compile-verified, matrix operand order checked by reading, not run on hardware
//
#include <hip/hip_runtime.h>
#include <hip/hip_bf16.h>

// ---------------------------------------------------------------------------
// Text->Img cross attention, CDNA5 (gfx1250, wave32) bf16 WMMA implementation
//   B=16, N=128, H=W=64 (M=4096), TC=768, IC=512, FQ=36 (padded to 64)
// ---------------------------------------------------------------------------

typedef __bf16 bf16;
typedef __attribute__((ext_vector_type(8)))  bf16  bf16x8;
typedef __attribute__((ext_vector_type(16))) bf16  bf16x16;
typedef __attribute__((ext_vector_type(8)))  float f32x8;

// A/B fragment loader for V_WMMA_*_16X16X32_BF16 (wave32).
// Per ISA 7.12.2, lane (with half = lane>>4) holds K = half*8 + [0..7] in
// elements 0..7 and K = 16 + half*8 + [0..7] in elements 8..15 -> two 16B
// contiguous loads from a row-major [row/col, K] buffer.
__device__ __forceinline__ bf16x16 load_frag(const bf16* base, int half) {
  bf16x8 lo = *(const bf16x8*)(base + half * 8);
  bf16x8 hi = *(const bf16x8*)(base + 16 + half * 8);
  bf16x16 r;
#pragma unroll
  for (int i = 0; i < 8; ++i) { r[i] = lo[i]; r[i + 8] = hi[i]; }
  return r;
}

__device__ __forceinline__ f32x8 wmma_bf16(bf16x16 a, bf16x16 b, f32x8 c) {
  // (neg_a, A, neg_b, B, c_mod, C, reuse_a, reuse_b)
  return __builtin_amdgcn_wmma_f32_16x16x32_bf16(false, a, false, b,
                                                 (short)0, c, false, false);
}

// Staging helpers for the projection GEMM (split so loads can be issued
// before the WMMAs and the cvt+LDS-store after -> VMEM latency hidden).
// A chunk: [128 rows x 32 k] f32, 2 threads/row x 16 k.  rows is always a
// multiple of 128 in our launches, so no row guard (keeps the K-loop free of
// divergent exec-mask save/restore).
__device__ __forceinline__ void stage_load(const float* __restrict__ A,
                                           const float* __restrict__ W,
                                           int rowblock, int colblock,
                                           int K, int ncols, int k0, int tid,
                                           float aR[16], float wR[8]) {
  const int r  = tid >> 1;
  const int kk = (tid & 1) * 16;
  const float* asrc = A + (size_t)(rowblock + r) * K + k0 + kk;
#pragma unroll
  for (int i = 0; i < 16; ++i) aR[i] = asrc[i];
  const int kw = tid >> 3;
  const int c0 = (tid & 7) * 8;
  const float* wsrc = W + (size_t)(k0 + kw) * ncols + colblock + c0;
#pragma unroll
  for (int i = 0; i < 8; ++i)
    wR[i] = (colblock + c0 + i < ncols) ? wsrc[i] : 0.f;  // zero-pad FQ->64
}

__device__ __forceinline__ void stage_store(bf16* __restrict__ lA,
                                            bf16* __restrict__ lB, int tid,
                                            const float aR[16], const float wR[8]) {
  const int r  = tid >> 1;
  const int kk = (tid & 1) * 16;
  bf16* dst = &lA[r * 32 + kk];
#pragma unroll
  for (int i = 0; i < 16; ++i) dst[i] = (bf16)aR[i];
  const int kw = tid >> 3;
  const int c0 = (tid & 7) * 8;
#pragma unroll
  for (int i = 0; i < 8; ++i) lB[(c0 + i) * 32 + kw] = (bf16)wR[i];  // [col][k]
}

// ---------------------------------------------------------------------------
// Kernel 1: out = relu(A[rows,K] @ W[K,ncols] + bias), f32 in -> bf16 out.
// Block = 256 thr (8 waves). Tile = 128 rows x 64 cols; wave w owns rows
// [16w,16w+16) x all 64 cols (4 accumulators). Double-buffered LDS staging:
// per K-chunk -> issue next chunk's global loads, run 4 WMMAs on current
// chunk (overlapping the loads), then cvt+store next chunk, one barrier.
// transpose=1 stores out as [b, col, m] (column-major per batch, rpb=2^rpbShift)
// for the o-GEMM's B operand; else row-major [row, out_ld] (zero-pad cols kept).
// ---------------------------------------------------------------------------
__launch_bounds__(256)
__global__ void proj_gemm_bias_relu(const float* __restrict__ A,
                                    const float* __restrict__ W,
                                    const float* __restrict__ bias,
                                    bf16* __restrict__ out,
                                    int rows, int K, int ncols,
                                    int out_ld, int transpose, int rpbShift) {
  __shared__ __align__(16) bf16 lA[2][128 * 32];  // [buf][row][k]  2x8 KB
  __shared__ __align__(16) bf16 lB[2][64 * 32];   // [buf][col][k]  2x4 KB

  const int tid  = threadIdx.x;
  const int w    = tid >> 5;
  const int lane = tid & 31;
  const int half = lane >> 4;
  const int l15  = lane & 15;
  const int rowblock = blockIdx.x * 128;
  const int colblock = blockIdx.y * 64;

  f32x8 acc[4] = {};
  float aR[16], wR[8];

  // prologue: stage chunk 0 into buffer 0
  stage_load(A, W, rowblock, colblock, K, ncols, 0, tid, aR, wR);
  stage_store(lA[0], lB[0], tid, aR, wR);
  __syncthreads();

  const int nCh = K >> 5;
  for (int ci = 0; ci < nCh; ++ci) {
    const int cur = ci & 1;
    const bool more = (ci + 1) < nCh;
    if (more)  // issue VMEM for next chunk before the matrix ops
      stage_load(A, W, rowblock, colblock, K, ncols, (ci + 1) * 32, tid, aR, wR);

    const bf16x16 af = load_frag(&lA[cur][(w * 16 + l15) * 32], half);
#pragma unroll
    for (int t = 0; t < 4; ++t) {
      const bf16x16 bfr = load_frag(&lB[cur][(t * 16 + l15) * 32], half);
      acc[t] = wmma_bf16(af, bfr, acc[t]);
    }

    if (more) {
      // safe: everyone finished reading buffer cur^1 before last barrier
      stage_store(lA[cur ^ 1], lB[cur ^ 1], tid, aR, wR);
      __syncthreads();
    }
  }

  // epilogue: bias + relu + bf16 store (C/D layout: row=r+8*half, col=l15)
#pragma unroll
  for (int t = 0; t < 4; ++t) {
    const int gc = colblock + t * 16 + l15;
    const float bv = (gc < ncols) ? bias[gc] : 0.f;
#pragma unroll
    for (int r = 0; r < 8; ++r) {
      const int grow = rowblock + w * 16 + r + 8 * half;  // rows % 128 == 0
      float val = acc[t][r] + bv;
      val = val > 0.f ? val : 0.f;
      if (transpose) {
        const int bb = grow >> rpbShift;
        const int mm = grow & ((1 << rpbShift) - 1);
        out[((size_t)bb * out_ld + gc) << rpbShift | (unsigned)mm] = (bf16)val;
      } else if (gc < out_ld) {
        out[(size_t)grow * out_ld + gc] = (bf16)val;
      }
    }
  }
}

// ---------------------------------------------------------------------------
// Kernel 2: s = q@k^T, beta = s*mask, two-pass softmax over M=4096.
// One block per (b, 16-query tile). Wave w owns m-tiles w, w+8, ... (32 each).
// Pass 1: WMMA scores (K=64 -> 2 wmma) -> beta f32 (ws) + running row max.
// Pass 2: exp(beta - max) -> bf16 p (unnormalized), row sums -> rowsum_g.
// Row reductions: __shfl_xor within 16-lane groups, then LDS across waves.
// ---------------------------------------------------------------------------
__launch_bounds__(256)
__global__ void attn_scores_softmax(const bf16* __restrict__ qp,
                                    const bf16* __restrict__ kp,
                                    const float* __restrict__ masks,
                                    float* __restrict__ beta,
                                    bf16* __restrict__ p,
                                    float* __restrict__ rowsum_g) {
  __shared__ float red[8][16];
  __shared__ float rowmax[16];

  const int tid  = threadIdx.x;
  const int w    = tid >> 5;
  const int lane = tid & 31;
  const int half = lane >> 4;
  const int l15  = lane & 15;
  const int b     = blockIdx.x >> 3;
  const int nbase = (blockIdx.x & 7) * 16;

  const bf16* qrow = qp + ((size_t)b * 128 + nbase + l15) * 64;
  const bf16x16 a0 = load_frag(qrow, half);        // K 0..31
  const bf16x16 a1 = load_frag(qrow + 32, half);   // K 32..63

  const size_t bm = (size_t)b * 128 * 4096;
  float rmax[8];
#pragma unroll
  for (int r = 0; r < 8; ++r) rmax[r] = -3.0e38f;

  for (int mt = w; mt < 256; mt += 8) {
    const int mbase = mt * 16;
    const bf16* krow = kp + ((size_t)b * 4096 + mbase + l15) * 64;
    const bf16x16 b0 = load_frag(krow, half);
    const bf16x16 b1 = load_frag(krow + 32, half);
    f32x8 acc = {};
    acc = wmma_bf16(a0, b0, acc);
    acc = wmma_bf16(a1, b1, acc);
    const int m = mbase + l15;
#pragma unroll
    for (int r = 0; r < 8; ++r) {
      const int n = nbase + r + 8 * half;
      const size_t idx = bm + (size_t)n * 4096 + m;
      const float bv = acc[r] * masks[idx];  // multiplicative mask BEFORE softmax
      beta[idx] = bv;
      rmax[r] = fmaxf(rmax[r], bv);
    }
  }

  // row max: reduce across the 16 lanes that share rows, then across waves
#pragma unroll
  for (int r = 0; r < 8; ++r)
#pragma unroll
    for (int off = 1; off < 16; off <<= 1)
      rmax[r] = fmaxf(rmax[r], __shfl_xor(rmax[r], off, 16));
  if (l15 == 0)
#pragma unroll
    for (int r = 0; r < 8; ++r) red[w][r + 8 * half] = rmax[r];
  __syncthreads();
  if (tid < 16) {
    float mx = -3.0e38f;
    for (int ww = 0; ww < 8; ++ww) mx = fmaxf(mx, red[ww][tid]);
    rowmax[tid] = mx;
  }
  __syncthreads();

  float rsum[8] = {};
  for (int mt = w; mt < 256; mt += 8) {
    const int m = mt * 16 + l15;
#pragma unroll
    for (int r = 0; r < 8; ++r) {
      const int n = nbase + r + 8 * half;
      const size_t idx = bm + (size_t)n * 4096 + m;
      const float e = __expf(beta[idx] - rowmax[r + 8 * half]);
      rsum[r] += e;
      p[idx] = (bf16)e;  // unnormalized; 1/sum applied in attn_out
    }
  }
#pragma unroll
  for (int r = 0; r < 8; ++r)
#pragma unroll
    for (int off = 1; off < 16; off <<= 1)
      rsum[r] += __shfl_xor(rsum[r], off, 16);
  __syncthreads();  // rowmax reads done; red is reusable
  if (l15 == 0)
#pragma unroll
    for (int r = 0; r < 8; ++r) red[w][r + 8 * half] = rsum[r];
  __syncthreads();
  if (tid < 16) {
    float s = 0.f;
    for (int ww = 0; ww < 8; ++ww) s += red[ww][tid];
    rowsum_g[(size_t)b * 128 + nbase + tid] = s;
  }
}

// ---------------------------------------------------------------------------
// Kernel 3: o[b,n,col] = (p[b,n,:] @ v[b,:,col]) / rowsum[b,n].
// v stored transposed [b, col, m] so B-fragments are K(m)-contiguous.
// Two independent accumulators over even/odd K-chunks hide WMMA RAW latency.
// ---------------------------------------------------------------------------
__launch_bounds__(256)
__global__ void attn_out(const bf16* __restrict__ p,
                         const bf16* __restrict__ v_t,
                         const float* __restrict__ rowsum_g,
                         float* __restrict__ out) {
  const int tid  = threadIdx.x;
  const int w    = tid >> 5;
  const int lane = tid & 31;
  const int half = lane >> 4;
  const int l15  = lane & 15;
  const int b     = blockIdx.x >> 3;
  const int nbase = (blockIdx.x & 7) * 16;
  const int col   = blockIdx.y * 128 + w * 16 + l15;

  const bf16* arow = p   + ((size_t)b * 128 + nbase + l15) * 4096;
  const bf16* brow = v_t + ((size_t)b * 768 + col) * 4096;

  f32x8 acc0 = {}, acc1 = {};
  for (int m0 = 0; m0 < 4096; m0 += 64) {
    if (m0 + 128 < 4096) {
      __builtin_prefetch(brow + m0 + 128, 0, 0);  // global_prefetch_b8
      __builtin_prefetch(arow + m0 + 128, 0, 0);
    }
    const bf16x16 af0 = load_frag(arow + m0, half);
    const bf16x16 bf0 = load_frag(brow + m0, half);
    acc0 = wmma_bf16(af0, bf0, acc0);
    const bf16x16 af1 = load_frag(arow + m0 + 32, half);
    const bf16x16 bf1 = load_frag(brow + m0 + 32, half);
    acc1 = wmma_bf16(af1, bf1, acc1);
  }
#pragma unroll
  for (int r = 0; r < 8; ++r) {
    const int n = nbase + r + 8 * half;
    const float scale = 1.0f / rowsum_g[(size_t)b * 128 + n];
    out[((size_t)b * 128 + n) * 768 + col] = (acc0[r] + acc1[r]) * scale;
  }
}

// ---------------------------------------------------------------------------
extern "C" void kernel_launch(void* const* d_in, const int* in_sizes, int n_in,
                              void* d_out, int out_size, void* d_ws, size_t ws_size,
                              hipStream_t stream) {
  (void)in_sizes; (void)n_in; (void)out_size; (void)ws_size;
  const float* text  = (const float*)d_in[0];  // [16,128,768]
  const float* img   = (const float*)d_in[1];  // [16,64,64,512] = [65536,512]
  const float* masks = (const float*)d_in[2];  // [16,128,4096]
  const float* Wq    = (const float*)d_in[3];  // [768,36]
  const float* bq    = (const float*)d_in[4];
  const float* Wk    = (const float*)d_in[5];  // [512,36]
  const float* bk    = (const float*)d_in[6];
  const float* Wv    = (const float*)d_in[7];  // [512,768]
  const float* bv    = (const float*)d_in[8];
  float* out = (float*)d_out;                  // [16,128,768]

  // workspace carve (256B aligned regions)
  char* ws = (char*)d_ws;
  size_t off = 0;
  auto carve = [&](size_t bytes) -> void* {
    void* ptr = ws + off;
    off = (off + bytes + 255) & ~(size_t)255;
    return ptr;
  };
  bf16*  qp    = (bf16*)carve((size_t)2048 * 64 * 2);          //  256 KB
  bf16*  kpB   = (bf16*)carve((size_t)65536 * 64 * 2);         //    8 MB
  bf16*  vt    = (bf16*)carve((size_t)16 * 768 * 4096 * 2);    //  100 MB
  bf16*  pP    = (bf16*)carve((size_t)16 * 128 * 4096 * 2);    //   16 MB
  float* beta  = (float*)carve((size_t)16 * 128 * 4096 * 4);   //   33 MB
  float* rsum  = (float*)carve((size_t)2048 * 4);

  // q = relu(text@Wq+bq) -> [2048,64] bf16 (cols 36..63 zero)
  proj_gemm_bias_relu<<<dim3(16, 1), 256, 0, stream>>>(
      text, Wq, bq, qp, 2048, 768, 36, 64, 0, 0);
  // k = relu(img@Wk+bk) -> [65536,64] bf16
  proj_gemm_bias_relu<<<dim3(512, 1), 256, 0, stream>>>(
      img, Wk, bk, kpB, 65536, 512, 36, 64, 0, 0);
  // v = relu(img@Wv+bv) -> transposed [16,768,4096] bf16 (rpb = 2^12)
  proj_gemm_bias_relu<<<dim3(512, 12), 256, 0, stream>>>(
      img, Wv, bv, vt, 65536, 512, 768, 768, 1, 12);
  // scores + masked softmax
  attn_scores_softmax<<<dim3(128), 256, 0, stream>>>(
      qp, kpB, masks, beta, pP, rsum);
  // o = (p @ v) / rowsum
  attn_out<<<dim3(128, 6), 256, 0, stream>>>(pP, vt, rsum, out);
}
